// RecurrentActorCritic_18373870092920
// MI455X (gfx1250) — matrix operand order repaired and checked by reading, not compiled, not run on hardware
//
#include <hip/hip_runtime.h>

typedef __attribute__((ext_vector_type(16))) _Float16 v16h;
typedef __attribute__((ext_vector_type(8)))  _Float16 v8h;
typedef __attribute__((ext_vector_type(8)))  float    v8f;
typedef __attribute__((ext_vector_type(4)))  unsigned v4u;
typedef __attribute__((ext_vector_type(8)))  unsigned v8u;

#define TT 256
#define BB 512
#define DD 64
#define HH 256
#define LL 128
#define AA 8
#define E1W 128   // layer-1 width

// workspace layout (units: _Float16 elements)
#define OFF_W1  0
#define SZ_W1   (DD * E1W)        // 8192
#define OFF_W2  (OFF_W1 + SZ_W1)
#define SZ_W2   (E1W * HH)        // 32768
#define OFF_WIH (OFF_W2 + SZ_W2)
#define SZ_WIH  (HH * 4 * LL)     // 131072
#define OFF_WHH (OFF_WIH + SZ_WIH)
#define SZ_WHH  (LL * 4 * LL)     // 65536
#define OFF_E   (OFF_WHH + SZ_WHH)
#define SZ_E    ((size_t)TT * BB * HH)

// LDS stride (in halfs) for the TDM-staged e tile: 256 halfs + 8-half pad
// (TDM pad: 16 B after every 512 B) -> 528 B row stride, breaks 64-bank wrap.
#define ELDA 264

// ---------------------------------------------------------------------------
// A-fragment (16x32 f16, ISA 7.12.2): lane holds row M=lane&15; halfs j<8 are
// K = koff+j, halfs j>=8 are K = koff+16+j-8, koff = (lane&16)?8:0.
// Two contiguous 16B loads per lane.
__device__ __forceinline__ v16h make_a(const _Float16* p) {
  v8h lo = *(const v8h*)p;
  v8h hi = *(const v8h*)(p + 16);
  return __builtin_shufflevector(lo, hi, 0, 1, 2, 3, 4, 5, 6, 7,
                                 8, 9, 10, 11, 12, 13, 14, 15);
}

__device__ __forceinline__ v8f wmma16(v16h a, v16h b, v8f c) {
  // (neg_a, A, neg_b, B, c_mod, C, reuse_a, reuse_b)
  return __builtin_amdgcn_wmma_f32_16x16x32_f16(false, a, false, b,
                                                (short)0, c, false, false);
}

__device__ __forceinline__ float fsigmoid(float x) {
  return 1.0f / (1.0f + __expf(-x));
}
__device__ __forceinline__ float ftanh(float x) {
  return 2.0f / (1.0f + __expf(-2.0f * x)) - 1.0f;
}

// ---------------------------------------------------------------------------
// Tensor Data Mover: DMA one contiguous 8 KB tile (16 rows x 256 f16) from
// global into LDS, inserting a 16 B pad after every 512 B (row) so LDS rows
// land at a 528 B stride. D# built per CDNA5 ISA.
//   group0: count=1 | lds_addr | global_addr | type=2
//   group1: data_size=2B, pad_enable, pad_interval=6 (64 qwords = 512 B),
//           pad_amount=3 (4 dwords = 16 B), tensor/tile dim0 = 4096 elements,
//           dim1 = 1, strides = 4096.
// Completion tracked with TENSORcnt.
__device__ __forceinline__ void tdm_load_tile(const _Float16* gsrc, void* ldst) {
  unsigned lds_lo = (unsigned)(uintptr_t)ldst;   // low 32 bits = LDS offset
  unsigned long long ga = (unsigned long long)(uintptr_t)gsrc;
  v4u g0;
  g0[0] = 1u;                                        // count = 1, load, user
  g0[1] = lds_lo;                                    // lds_addr[63:32]
  g0[2] = (unsigned)(ga & 0xffffffffu);              // global_addr[95:64]
  g0[3] = (unsigned)((ga >> 32) & 0x01ffffffu)       // global_addr[120:96]
        | 0x80000000u;                               // type[127:126] = 2
  v8u g1;
  g1[0] = 0x07910000u;   // data_size=1(2B) | pad_enable | interval=6 | amount=3
  g1[1] = 0x10000000u;   // tensor_dim0 = 4096 (bits[79:48], low half here)
  g1[2] = 0x00010000u;   // tensor_dim0 hi = 0, tensor_dim1 = 1
  g1[3] = 0x10000000u;   // tensor_dim1 hi = 0, tile_dim0 = 4096
  g1[4] = 0x00000001u;   // tile_dim1 = 1, tile_dim2 = 0
  g1[5] = 4096u;         // tensor_dim0_stride low32
  g1[6] = 0x10000000u;   // dim0_stride hi = 0, tensor_dim1_stride low16 = 4096
  g1[7] = 0u;            // tensor_dim1_stride hi
  asm volatile("tensor_load_to_lds %0, %1" :: "s"(g0), "s"(g1) : "memory");
}

// ---------------------------------------------------------------------------
// Pack W [N][K] row-major (gemm computes X @ W^T) into WMMA B-fragment order:
// tiles (nt, kt) of 32K x 16N; per tile 32 lanes x 16 halfs contiguous.
// B-fragment (32x16 f16): lane holds col N = lane&15, half j<8 is
// K = koff + j, half j>=8 is K = koff + 16 + j-8, koff = (lane&16)?8:0.
__global__ void pack_w(const float* __restrict__ W, _Float16* __restrict__ out,
                       int K, int N) {
  int idx = blockIdx.x * blockDim.x + threadIdx.x;
  if (idx >= K * N) return;
  int j    = idx & 15;
  int lane = (idx >> 4) & 31;
  int tile = idx >> 9;
  int KT   = K >> 5;
  int kt   = tile % KT;
  int nt   = tile / KT;
  int k = kt * 32 + ((lane & 16) ? 8 : 0) + ((j < 8) ? j : 16 + j - 8);
  int n = nt * 16 + (lane & 15);
  out[idx] = (_Float16)W[n * K + k];
}

// ---------------------------------------------------------------------------
// Encoder: e = relu(relu(clip(norm(obs)) @ W1^T + b1) @ W2^T + b2)
// 128 rows per block, 8 waves, each wave owns a 16-row strip.
__global__ __launch_bounds__(256)
void encoder_kernel(const float* __restrict__ obs,
                    const float* __restrict__ mean,
                    const float* __restrict__ var,
                    const _Float16* __restrict__ w1p,
                    const float* __restrict__ b1,
                    const _Float16* __restrict__ w2p,
                    const float* __restrict__ b2,
                    _Float16* __restrict__ e) {
  __shared__ __align__(16) _Float16 sX[128][DD + 8];
  __shared__ __align__(16) _Float16 sE1[128][E1W + 8];
  const int tid  = threadIdx.x;
  const int lane = tid & 31;
  const int wv   = tid >> 5;
  const long rowBase = (long)blockIdx.x * 128;

  // stage 1: normalize + clip -> f16 LDS tile
  for (int i = tid; i < 128 * DD; i += 256) {
    int r = i >> 6, d = i & (DD - 1);
    float vv = var[d];
    vv = vv < 1e-6f ? 1e-6f : vv;
    float x = (obs[(rowBase + r) * DD + d] - mean[d]) * rsqrtf(vv);
    x = fminf(10.0f, fmaxf(-10.0f, x));
    sX[r][d] = (_Float16)x;
  }
  __syncthreads();

  const int m0   = wv * 16;
  const int rA   = m0 + (lane & 15);
  const int koff = (lane & 16) ? 8 : 0;
  const int mrow = m0 + ((lane & 16) ? 8 : 0);

  // stage 2: layer 1 (K=64, N=128)
  for (int nt = 0; nt < E1W / 16; ++nt) {
    v8f acc = {};
#pragma unroll
    for (int kt = 0; kt < DD / 32; ++kt) {
      v16h a = make_a(&sX[rA][kt * 32 + koff]);
      v16h b = *(const v16h*)&w1p[((nt * (DD / 32) + kt) * 32 + lane) * 16];
      acc = wmma16(a, b, acc);
    }
    int n = nt * 16 + (lane & 15);
    float bias = b1[n];
#pragma unroll
    for (int v = 0; v < 8; ++v) {
      float y = acc[v] + bias;
      sE1[mrow + v][n] = (_Float16)(y > 0.0f ? y : 0.0f);
    }
  }
  __syncthreads();

  // stage 3: layer 2 (K=128, N=256) -> global f16 [T*B][H]
  for (int nt = 0; nt < HH / 16; ++nt) {
    v8f acc = {};
#pragma unroll
    for (int kt = 0; kt < E1W / 32; ++kt) {
      v16h a = make_a(&sE1[rA][kt * 32 + koff]);
      v16h b = *(const v16h*)&w2p[((nt * (E1W / 32) + kt) * 32 + lane) * 16];
      acc = wmma16(a, b, acc);
    }
    int n = nt * 16 + (lane & 15);
    float bias = b2[n];
#pragma unroll
    for (int v = 0; v < 8; ++v) {
      float y = acc[v] + bias;
      e[(rowBase + mrow + v) * HH + n] = (_Float16)(y > 0.0f ? y : 0.0f);
    }
  }
}

// ---------------------------------------------------------------------------
// LSTM scan + heads. 16 batch rows per block (32 blocks), h/c resident in LDS.
// e[t] tiles are double-buffered in LDS via TDM (tensor_load_to_lds), so the
// per-step gate GEMM reads A operands purely from LDS and B operands from the
// L2-resident packed weights.
__global__ __launch_bounds__(256)
void lstm_kernel(const _Float16* __restrict__ e,
                 const _Float16* __restrict__ wihp,
                 const _Float16* __restrict__ whhp,
                 const float* __restrict__ h0,
                 const float* __restrict__ c0,
                 const int* __restrict__ done,
                 const int* __restrict__ amask,
                 const float* __restrict__ b_ih,
                 const float* __restrict__ b_hh,
                 const float* __restrict__ Wp,
                 const float* __restrict__ bp,
                 const float* __restrict__ Wv,
                 const float* __restrict__ bv,
                 float* __restrict__ out_logits,
                 float* __restrict__ out_values,
                 float* __restrict__ out_h,
                 float* __restrict__ out_c) {
  __shared__ float sH[16][LL];
  __shared__ float sC[16][LL];
  __shared__ __align__(16) _Float16 sH16[16][LL + 8];
  __shared__ float sG[16][4 * LL + 8];
  __shared__ float sWpv[9][LL];
  __shared__ float sBias[4 * LL];
  __shared__ __align__(16) _Float16 sE[2][16][ELDA];  // TDM double buffer

  const int tid  = threadIdx.x;
  const int lane = tid & 31;
  const int wv   = tid >> 5;
  const int row0 = blockIdx.x * 16;

  for (int i = tid; i < 16 * LL; i += 256) {
    int r = i >> 7, j = i & (LL - 1);
    sH[r][j] = h0[(row0 + r) * LL + j];
    sC[r][j] = c0[(row0 + r) * LL + j];
  }
  for (int i = tid; i < 9 * LL; i += 256) {
    int o = i >> 7, j = i & (LL - 1);
    sWpv[o][j] = (o < 8) ? Wp[o * LL + j] : Wv[j];
  }
  for (int i = tid; i < 4 * LL; i += 256) sBias[i] = b_ih[i] + b_hh[i];

  // prime the pipeline: DMA e[0] tile into buffer 0
  if (wv == 0) {
    tdm_load_tile(&e[(size_t)row0 * HH], &sE[0][0][0]);
  }
  __syncthreads();

  const int koff = (lane & 16) ? 8 : 0;
  const int rA   = lane & 15;
  const int mrow = (lane & 16) ? 8 : 0;

  for (int t = 0; t < TT; ++t) {
    // 1: episode-reset scaling of (h, c); h -> f16 A-operand
    for (int i = tid; i < 16 * LL; i += 256) {
      int r = i >> 7, j = i & (LL - 1);
      float rt = 1.0f - (float)done[t * BB + row0 + r];
      float hh = sH[r][j] * rt;
      sC[r][j] = sC[r][j] * rt;
      sH16[r][j] = (_Float16)hh;
    }
    if (wv == 0) __builtin_amdgcn_s_wait_tensorcnt(0);  // e[t] tile landed
    __syncthreads();

    // kick off DMA of next step's tile into the other buffer
    if (wv == 0 && t + 1 < TT) {
      tdm_load_tile(&e[((size_t)(t + 1) * BB + row0) * HH],
                    &sE[(t + 1) & 1][0][0]);
    }

    // 2: gate GEMM (A from LDS, B from packed weights in L2)
    const _Float16* eRow = &sE[t & 1][rA][0];
    for (int i = 0; i < 4; ++i) {
      int ntg = wv * 4 + i;
      v8f acc = {};
#pragma unroll
      for (int kt = 0; kt < HH / 32; ++kt) {           // input contribution
        v16h a = make_a(eRow + kt * 32 + koff);
        v16h b = *(const v16h*)&wihp[((ntg * (HH / 32) + kt) * 32 + lane) * 16];
        acc = wmma16(a, b, acc);
      }
#pragma unroll
      for (int kt = 0; kt < LL / 32; ++kt) {           // recurrent contribution
        v16h a = make_a(&sH16[rA][kt * 32 + koff]);
        v16h b = *(const v16h*)&whhp[((ntg * (LL / 32) + kt) * 32 + lane) * 16];
        acc = wmma16(a, b, acc);
      }
      int n = ntg * 16 + (lane & 15);
#pragma unroll
      for (int v = 0; v < 8; ++v) sG[mrow + v][n] = acc[v];
    }
    __syncthreads();

    // 3: elementwise LSTM cell (gate order i,f,g,o)
    for (int i = tid; i < 16 * LL; i += 256) {
      int r = i >> 7, j = i & (LL - 1);
      float gi = sG[r][j]          + sBias[j];
      float gf = sG[r][LL + j]     + sBias[LL + j];
      float gg = sG[r][2 * LL + j] + sBias[2 * LL + j];
      float go = sG[r][3 * LL + j] + sBias[3 * LL + j];
      float c = fsigmoid(gf) * sC[r][j] + fsigmoid(gi) * ftanh(gg);
      float h = fsigmoid(go) * ftanh(c);
      sC[r][j] = c;
      sH[r][j] = h;
    }
    __syncthreads();

    // 4: policy/value heads (A=8 logits + 1 value per row)
    if (tid < 16 * 9) {
      int r = tid / 9, o = tid % 9;
      int b = row0 + r;
      float dot = 0.0f;
#pragma unroll 4
      for (int j = 0; j < LL; ++j) dot += sH[r][j] * sWpv[o][j];
      size_t base = (size_t)t * BB + b;
      if (o < 8) {
        float lg = dot + bp[o];
        out_logits[base * AA + o] = (amask[base * AA + o] > 0) ? lg : -1e9f;
      } else {
        out_values[base] = dot + bv[0];
      }
    }
    __syncthreads();
  }

  for (int i = tid; i < 16 * LL; i += 256) {
    int r = i >> 7, j = i & (LL - 1);
    out_h[(row0 + r) * LL + j] = sH[r][j];
    out_c[(row0 + r) * LL + j] = sC[r][j];
  }
}

// ---------------------------------------------------------------------------
extern "C" void kernel_launch(void* const* d_in, const int* in_sizes, int n_in,
                              void* d_out, int out_size, void* d_ws, size_t ws_size,
                              hipStream_t stream) {
  (void)in_sizes; (void)n_in; (void)out_size; (void)ws_size;

  const float* obs   = (const float*)d_in[0];
  const float* h0    = (const float*)d_in[1];
  const float* c0    = (const float*)d_in[2];
  const int*   done  = (const int*)d_in[3];
  const int*   amask = (const int*)d_in[4];
  const float* mean  = (const float*)d_in[5];
  const float* var   = (const float*)d_in[6];
  const float* W1    = (const float*)d_in[7];
  const float* b1    = (const float*)d_in[8];
  const float* W2    = (const float*)d_in[9];
  const float* b2    = (const float*)d_in[10];
  const float* W_ih  = (const float*)d_in[11];
  const float* W_hh  = (const float*)d_in[12];
  const float* b_ih  = (const float*)d_in[13];
  const float* b_hh  = (const float*)d_in[14];
  const float* Wp    = (const float*)d_in[15];
  const float* bp    = (const float*)d_in[16];
  const float* Wv    = (const float*)d_in[17];
  const float* bv    = (const float*)d_in[18];

  _Float16* ws   = (_Float16*)d_ws;
  _Float16* w1p  = ws + OFF_W1;
  _Float16* w2p  = ws + OFF_W2;
  _Float16* wihp = ws + OFF_WIH;
  _Float16* whhp = ws + OFF_WHH;
  _Float16* ebuf = ws + OFF_E;

  float* out_logits = (float*)d_out;
  float* out_values = out_logits + (size_t)TT * BB * AA;
  float* out_h      = out_values + (size_t)TT * BB;
  float* out_c      = out_h + (size_t)BB * LL;

  pack_w<<<(SZ_W1  + 255) / 256, 256, 0, stream>>>(W1,   w1p,  DD,  E1W);
  pack_w<<<(SZ_W2  + 255) / 256, 256, 0, stream>>>(W2,   w2p,  E1W, HH);
  pack_w<<<(SZ_WIH + 255) / 256, 256, 0, stream>>>(W_ih, wihp, HH,  4 * LL);
  pack_w<<<(SZ_WHH + 255) / 256, 256, 0, stream>>>(W_hh, whhp, LL,  4 * LL);

  encoder_kernel<<<(TT * BB) / 128, 256, 0, stream>>>(obs, mean, var,
                                                      w1p, b1, w2p, b2, ebuf);

  lstm_kernel<<<BB / 16, 256, 0, stream>>>(ebuf, wihp, whhp, h0, c0, done, amask,
                                           b_ih, b_hh, Wp, bp, Wv, bv,
                                           out_logits, out_values, out_h, out_c);
}
